// NGMCDR_90718299226375
// MI455X (gfx1250) — matrix operand order repaired and checked by reading, not compiled, not run on hardware
//
#include <hip/hip_runtime.h>
#include <hip/hip_bf16.h>

// ---------------------------------------------------------------------------
// NGMCDR forward for MI455X (gfx1250, wave32, WMMA).
// GEMMs via v_wmma_f32_16x16x32_f16 with pre-packed (WMMA-ready) f16 operands:
// each lane's 16-half fragment is contiguous -> b128 vector loads, no packing
// VALU in the hot loop. Epilogue flags are template params (no runtime
// pointer-null branches in the kernel).
// ---------------------------------------------------------------------------

#define NN  512
#define DD  64
#define LL  8
#define NLD 4096   // N*L
#define HH  128

typedef __attribute__((ext_vector_type(16))) _Float16 v16h;
typedef __attribute__((ext_vector_type(8)))  float    v8f;

// ------------------------------ WMMA GEMM ----------------------------------
// Packed A layout: [mt][kt][lane][e0..15], fragment in contiguous k order (not the ISA table order):
//   row = mt*16 + (lane&15), k = kt*32 + e + (lane>>4)*16
// Packed B layout: [nt][kt][lane][e0..15]:
//   col = nt*16 + (lane&15), k = kt*32 + e + (lane>>4)*16
// Each wave computes a 16x32 C tile (2 accumulators, shared A fragment).
template <bool BIAS, bool RES, bool RSC, bool RELU>
__global__ __launch_bounds__(32) void gemm_wmma_k(
    const _Float16* __restrict__ Ap, const _Float16* __restrict__ Bp,
    float* __restrict__ C,
    const float* __restrict__ bias, const float* __restrict__ residual,
    const float* __restrict__ rowscale, float alpha, int Kt, int Nc) {
  const int ntp = blockIdx.x;          // 32-column tile pair
  const int mt  = blockIdx.y;
  const int lane = threadIdx.x;
  const int hi = lane >> 4;
  const int mr = lane & 15;

  const v16h* Aq = ((const v16h*)Ap) + (size_t)mt * Kt * 32 + lane;
  const v16h* B0 = ((const v16h*)Bp) + (size_t)(2 * ntp) * Kt * 32 + lane;
  const v16h* B1 = B0 + (size_t)Kt * 32;

  v8f acc0 = {}, acc1 = {};
#pragma unroll 2
  for (int kt = 0; kt < Kt; ++kt) {
    v16h a  = Aq[(size_t)kt * 32];
    v16h b0 = B0[(size_t)kt * 32];
    v16h b1 = B1[(size_t)kt * 32];
    acc0 = __builtin_amdgcn_wmma_f32_16x16x32_f16(
        false, a, false, b0, (short)0, acc0, false, false);
    acc1 = __builtin_amdgcn_wmma_f32_16x16x32_f16(
        false, a, false, b1, (short)0, acc1, false, false);
  }

  const int c0 = ntp * 32 + mr;
  const int c1 = c0 + 16;
#pragma unroll
  for (int v = 0; v < 8; ++v) {
    int row = mt * 16 + v + hi * 8;
    float v0 = acc0[v] * alpha;
    float v1 = acc1[v] * alpha;
    if (RSC)  { float r = rowscale[row]; v0 *= r; v1 *= r; }
    if (BIAS) { v0 += bias[c0]; v1 += bias[c1]; }
    if (RES)  { v0 += residual[(size_t)row * Nc + c0];
                v1 += residual[(size_t)row * Nc + c1]; }
    if (RELU) { v0 = fmaxf(v0, 0.0f); v1 = fmaxf(v1, 0.0f); }
    C[(size_t)row * Nc + c0] = v0;
    C[(size_t)row * Nc + c1] = v1;
  }
}

// ------------------------------ pack kernels -------------------------------
// f32 [M x K] row-major -> packed-A f16 (layout above). n = M*K threads.
__global__ void pack_a_k(const float* __restrict__ src, _Float16* __restrict__ dst,
                         int K, int Kt) {
  int q = blockIdx.x * 256 + threadIdx.x;
  int e = q & 15, lane = (q >> 4) & 31, kt = (q >> 9) % Kt, mt = q / (Kt << 9);
  int hi = lane >> 4, mr = lane & 15;
  int row = mt * 16 + mr;
  int k = kt * 32 + e + hi * 16;
  dst[q] = (_Float16)src[(size_t)row * K + k];
}

// f32 [K x Nc] row-major -> packed-B f16. n = K*Nc threads.
__global__ void pack_b_k(const float* __restrict__ src, _Float16* __restrict__ dst,
                         int Nc, int Kt) {
  int q = blockIdx.x * 256 + threadIdx.x;
  int e = q & 15, lane = (q >> 4) & 31, kt = (q >> 9) % Kt, nt = q / (Kt << 9);
  int hi = lane >> 4, mr = lane & 15;
  int k = kt * 32 + e + hi * 16;
  int col = nt * 16 + mr;
  dst[q] = (_Float16)src[(size_t)k * Nc + col];
}

// f32 src [Nc x K] row-major, B = src^T -> packed-B f16 (fused transpose).
__global__ void pack_bt_k(const float* __restrict__ src, _Float16* __restrict__ dst,
                          int K, int Kt) {
  int q = blockIdx.x * 256 + threadIdx.x;
  int e = q & 15, lane = (q >> 4) & 31, kt = (q >> 9) % Kt, nt = q / (Kt << 9);
  int hi = lane >> 4, mr = lane & 15;
  int k = kt * 32 + e + hi * 16;
  int col = nt * 16 + mr;
  dst[q] = (_Float16)src[(size_t)col * K + k];
}

// ------------------------------ small kernels ------------------------------
// out[r, :] = table[idx[r], :]   (grid=rows, block=64)
__global__ void gather_k(const float* __restrict__ tab, const int* __restrict__ idx,
                         float* __restrict__ out) {
  int r = blockIdx.x, d = threadIdx.x;
  out[(size_t)r * DD + d] = tab[(size_t)idx[r] * DD + d];
}

// out[i,d] = mean_l s[(i*L+l), d]
__global__ void mean_seq_k(const float* __restrict__ s, float* __restrict__ out) {
  int i = blockIdx.x, d = threadIdx.x;
  float acc = 0.0f;
#pragma unroll
  for (int l = 0; l < LL; ++l) acc += s[((size_t)i * LL + l) * DD + d];
  out[(size_t)i * DD + d] = acc * (1.0f / LL);
}

// zero diag + L1-normalize each row of att [NN x NN] in place
__global__ void l1norm_zdiag_k(float* __restrict__ att) {
  int i = blockIdx.x, t = threadIdx.x;   // block 256
  __shared__ float red[256];
  float s = 0.0f;
  for (int j = t; j < NN; j += 256) {
    float v = (j == i) ? 0.0f : att[(size_t)i * NN + j];
    s += fabsf(v);
  }
  red[t] = s; __syncthreads();
  for (int k = 128; k > 0; k >>= 1) { if (t < k) red[t] += red[t + k]; __syncthreads(); }
  float inv = 1.0f / fmaxf(red[0], 1e-12f);
  for (int j = t; j < NN; j += 256) {
    float v = (j == i) ? 0.0f : att[(size_t)i * NN + j];
    att[(size_t)i * NN + j] = v * inv;
  }
}

// rowscale[i] = extra / max(sum_j |A[i,j]|, 1e-12)
__global__ void rowscale_abs_k(const float* __restrict__ A, float* __restrict__ rs,
                               int cols, float extra) {
  int i = blockIdx.x, t = threadIdx.x;   // block 256
  __shared__ float red[256];
  float s = 0.0f;
  for (int j = t; j < cols; j += 256) s += fabsf(A[(size_t)i * cols + j]);
  red[t] = s; __syncthreads();
  for (int k = 128; k > 0; k >>= 1) { if (t < k) red[t] += red[t + k]; __syncthreads(); }
  if (t == 0) rs[i] = extra / fmaxf(red[0], 1e-12f);
}

// gbias[d] = b_t1[d] + b_t2[d] + sum_k bl[k]*Wt1[k,d] + bt[k]*Wt2[k,d]
__global__ void gbias_k(const float* __restrict__ bl, const float* __restrict__ w1,
                        const float* __restrict__ b1, const float* __restrict__ bt,
                        const float* __restrict__ w2, const float* __restrict__ b2,
                        float* __restrict__ out) {
  int d = threadIdx.x;  // 64
  float s = b1[d] + b2[d];
  for (int k = 0; k < DD; ++k)
    s += bl[k] * w1[k * DD + d] + bt[k] * w2[k * DD + d];
  out[d] = s;
}

// m2 inner-layer reduction (per-pair sigmoid gate, mean over j):
__global__ void inner_reduce_k(float* __restrict__ u, const float* __restrict__ att,
                               const float* __restrict__ mask,
                               const float* __restrict__ vL, const float* __restrict__ vT,
                               const float* __restrict__ pL, const float* __restrict__ pT,
                               const float* __restrict__ gbias,
                               const float* __restrict__ bl, const float* __restrict__ bt) {
  int i = blockIdx.x, d = threadIdx.x;   // grid NN, block DD
  __shared__ float satt[NN], smask[NN];
  for (int j = d; j < NN; j += DD) {
    satt[j] = att[(size_t)i * NN + j];
    smask[j] = mask[j];
  }
  __syncthreads();
  float gb = gbias[d], b0 = bl[d], b1v = bt[d];
  float acc = 0.0f;
  for (int j = 0; j < NN; ++j) {
    float a = satt[j], m = smask[j];
    float cL = a * m, cT = a - cL;
    float x = cL * pL[(size_t)j * DD + d] + cT * pT[(size_t)j * DD + d] + gb;
    float g = 1.0f / (1.0f + __expf(-x));
    acc += g * (cL * vL[(size_t)j * DD + d] + b0)
         + (1.0f - g) * (cT * vT[(size_t)j * DD + d] + b1v);
  }
  int idx = i * DD + d;
  u[idx] = fmaxf(u[idx] + acc * (1.0f / NN), 0.0f);
}

// u += g*md + (1-g)*sd, g = sigmoid(ga+gb)
__global__ void gatefuse_k(float* __restrict__ u, const float* __restrict__ md,
                           const float* __restrict__ sd, const float* __restrict__ ga,
                           const float* __restrict__ gb, int n) {
  int i = blockIdx.x * blockDim.x + threadIdx.x;
  if (i >= n) return;
  float g = 1.0f / (1.0f + __expf(-(ga[i] + gb[i])));
  u[i] += g * md[i] + (1.0f - g) * sd[i];
}

__global__ void relu_k(float* __restrict__ x, int n) {
  int i = blockIdx.x * blockDim.x + threadIdx.x;
  if (i < n) x[i] = fmaxf(x[i], 0.0f);
}

// Build predictor input X [2048 x 128] (f32, packed later).
__global__ void build_x_k(const float* __restrict__ u1, const float* __restrict__ u2,
                          const float* __restrict__ emb_item,
                          const int* __restrict__ i_node, const int* __restrict__ negs,
                          float* __restrict__ X) {
  int r = blockIdx.x, c = threadIdx.x;   // 2048 x 128
  int pass = r >> 10, rem = r & 1023, i = rem >> 1, k = rem & 1;
  float v;
  if (c < DD) {
    v = (pass == 0 ? u1 : u2)[(size_t)i * DD + c];
  } else {
    int item = (k == 0) ? i_node[i] : negs[i];
    v = emb_item[(size_t)item * DD + (c - DD)];
  }
  X[(size_t)r * 2 * DD + c] = v;
}

// out[r] = sigmoid(dot(H[r,:], w2) + b2)
__global__ void fc2_sig_k(const float* __restrict__ H, const float* __restrict__ w2,
                          const float* __restrict__ b2, float* __restrict__ out) {
  int r = blockIdx.x, t = threadIdx.x;   // block 128
  __shared__ float red[HH];
  red[t] = H[(size_t)r * HH + t] * w2[t];
  __syncthreads();
  for (int k = 64; k > 0; k >>= 1) { if (t < k) red[t] += red[t + k]; __syncthreads(); }
  if (t == 0) out[r] = 1.0f / (1.0f + __expf(-(red[0] + b2[0])));
}

// ------------------------------ host orchestration -------------------------
extern "C" void kernel_launch(void* const* d_in, const int* in_sizes, int n_in,
                              void* d_out, int out_size, void* d_ws, size_t ws_size,
                              hipStream_t stream) {
  (void)in_sizes; (void)n_in; (void)out_size; (void)ws_size;
  const int*   u_node   = (const int*)d_in[0];
  const int*   i_node   = (const int*)d_in[1];
  const int*   negs     = (const int*)d_in[2];
  const int*   seq1     = (const int*)d_in[3];
  const int*   seq2     = (const int*)d_in[4];
  const float* mask1    = (const float*)d_in[5];
  const float* mask2    = (const float*)d_in[6];
  const float* emb_item = (const float*)d_in[7];
  const float* emb_user = (const float*)d_in[8];
  auto P = [&](int i) { return (const float*)d_in[i]; };
  // params (sorted-key pytree): 9:m1d1.b 10:m1d1.w 11:m1d2.b 12:m1d2.w
  // m2_d1 13-20: gf.t1.b/w, gf.t2.b/w, long.b/w, tail.b/w ; m2_d2 21-28 same
  // m3 29-44: gf1.t1.b/w gf1.t2.b/w gf2.t1.b/w gf2.t2.b/w sd1.b/w sd2.b/w td1.b/w td2.b/w
  // 45-48: m4d1.b/w m4d2.b/w ; 49-52: fc1.b/w fc2.b/w ; 53-56: transd1.b/w transd2.b/w

  float* fbase = (float*)d_ws;
  size_t o = 0;
  auto AF = [&](size_t n) { float* p = fbase + o; o += (n + 63) & ~size_t(63); return p; };
  float* ue   = AF(NN * DD);
  float* u1   = AF(NN * DD);
  float* u2   = AF(NN * DD);
  float* s1f  = AF(NLD * DD);
  float* s2f  = AF(NLD * DD);
  float* s1m  = AF(NN * DD);
  float* s2m  = AF(NN * DD);
  float* att  = AF(NN * NN);
  float* attL = AF((size_t)NN * NLD);
  float* tA   = AF(NN * DD);
  float* tB   = AF(NN * DD);
  float* tC   = AF(NN * DD);
  float* tD   = AF(NN * DD);
  float* tE   = AF(NN * DD);
  float* gbias = AF(DD);
  float* rsc  = AF(NN);
  float* tBig = AF(NLD * DD);
  float* Hprd = AF(2048 * HH);
  float* fX   = AF(2048 * HH);
  _Float16* hA = (_Float16*)(fbase + o);            // 512*4096 halfs max
  _Float16* hB = hA + (size_t)NN * NLD;             // 64*4096 halfs max
  _Float16* hW = hB + (size_t)DD * NLD;             // 128*128 halfs max

  auto packA = [&](const float* s, _Float16* d, int M, int K) {
    pack_a_k<<<(M * K) / 256, 256, 0, stream>>>(s, d, K, K / 32);
  };
  auto packB = [&](const float* s, _Float16* d, int K, int Nc) {
    pack_b_k<<<(K * Nc) / 256, 256, 0, stream>>>(s, d, Nc, K / 32);
  };
  auto packBT = [&](const float* s /* Nc x K */, _Float16* d, int K, int Nc) {
    pack_bt_k<<<(K * Nc) / 256, 256, 0, stream>>>(s, d, K, K / 32);
  };
  auto gemm = [&](const _Float16* A, const _Float16* B, float* C,
                  const float* bias, const float* resid, const float* rowsc,
                  float alpha, int relu, int M, int K, int Nc) {
    dim3 grid(Nc / 32, M / 16);
    int Kt = K / 32;
    if (rowsc)
      gemm_wmma_k<true, true, true, true><<<grid, 32, 0, stream>>>(A, B, C, bias, resid, rowsc, alpha, Kt, Nc);
    else if (resid)
      gemm_wmma_k<true, true, false, true><<<grid, 32, 0, stream>>>(A, B, C, bias, resid, rowsc, alpha, Kt, Nc);
    else if (bias && relu)
      gemm_wmma_k<true, false, false, true><<<grid, 32, 0, stream>>>(A, B, C, bias, resid, rowsc, alpha, Kt, Nc);
    else if (bias)
      gemm_wmma_k<true, false, false, false><<<grid, 32, 0, stream>>>(A, B, C, bias, resid, rowsc, alpha, Kt, Nc);
    else
      gemm_wmma_k<false, false, false, false><<<grid, 32, 0, stream>>>(A, B, C, bias, resid, rowsc, alpha, Kt, Nc);
  };

  // ---- stage 0: gathers + seq means ----
  gather_k<<<NN, DD, 0, stream>>>(emb_user, u_node, ue);
  gather_k<<<NLD, DD, 0, stream>>>(emb_item, seq1, s1f);
  gather_k<<<NLD, DD, 0, stream>>>(emb_item, seq2, s2f);
  mean_seq_k<<<NN, DD, 0, stream>>>(s1f, s1m);
  mean_seq_k<<<NN, DD, 0, stream>>>(s2f, s2m);

  // ---- stage 1: trans + m1 ----
  packA(ue, hA, NN, DD);
  packB(P(54), hW, DD, DD);
  gemm(hA, hW, u1, P(53), nullptr, nullptr, 1.0f, 0, NN, DD, DD);
  packB(P(56), hW, DD, DD);
  gemm(hA, hW, u2, P(55), nullptr, nullptr, 1.0f, 0, NN, DD, DD);
  packA(s1m, hA, NN, DD); packB(P(10), hW, DD, DD);
  gemm(hA, hW, u1, P(9), u1, nullptr, 1.0f, 1, NN, DD, DD);
  packA(s2m, hA, NN, DD); packB(P(12), hW, DD, DD);
  gemm(hA, hW, u2, P(11), u2, nullptr, 1.0f, 1, NN, DD, DD);

  // ---- stage 2: m2 inner layers ----
  auto inner = [&](float* u, const float* mask, int t1b, int t1w, int t2b, int t2w,
                   int lb, int lw, int tb, int tw) {
    packA(u, hA, NN, DD);
    packBT(u, hB, DD, NN);   // B = u^T
    gemm(hA, hB, att, nullptr, nullptr, nullptr, 1.0f, 0, NN, DD, NN);
    l1norm_zdiag_k<<<NN, 256, 0, stream>>>(att);
    packB(P(lw), hW, DD, DD);
    gemm(hA, hW, tA, nullptr, nullptr, nullptr, 1.0f, 0, NN, DD, DD);  // vL0
    packB(P(tw), hW, DD, DD);
    gemm(hA, hW, tB, nullptr, nullptr, nullptr, 1.0f, 0, NN, DD, DD);  // vT0
    packA(tA, hA, NN, DD); packB(P(t1w), hW, DD, DD);
    gemm(hA, hW, tC, nullptr, nullptr, nullptr, 1.0f, 0, NN, DD, DD);  // pL
    packA(tB, hA, NN, DD); packB(P(t2w), hW, DD, DD);
    gemm(hA, hW, tD, nullptr, nullptr, nullptr, 1.0f, 0, NN, DD, DD);  // pT
    gbias_k<<<1, DD, 0, stream>>>(P(lb), P(t1w), P(t1b), P(tb), P(t2w), P(t2b), gbias);
    inner_reduce_k<<<NN, DD, 0, stream>>>(u, att, mask, tA, tB, tC, tD, gbias, P(lb), P(tb));
  };
  inner(u1, mask1, 13, 14, 15, 16, 17, 18, 19, 20);
  inner(u2, mask2, 21, 22, 23, 24, 25, 26, 27, 28);

  // ---- stage 3: m3 inter layer ----
  auto inter_half = [&](float* ud, const float* us, int tdb, int tdw, int sdb, int sdw,
                        int g1b, int g1w, int g2b, int g2w) {
    packA(ud, hA, NN, DD);
    packBT(us, hB, DD, NN);
    gemm(hA, hB, att, nullptr, nullptr, nullptr, 1.0f, 0, NN, DD, NN);
    l1norm_zdiag_k<<<NN, 256, 0, stream>>>(att);
    packA(us, hA, NN, DD); packB(P(tdw), hW, DD, DD);
    gemm(hA, hW, tA, nullptr, nullptr, nullptr, 1.0f, 0, NN, DD, DD);   // v = us@Wtd
    packA(att, hA, NN, NN); packB(tA, hB, NN, DD);
    gemm(hA, hB, tB, P(tdb), nullptr, nullptr, 1.0f / NN, 0, NN, NN, DD);  // md
    packA(us, hA, NN, DD); packB(P(sdw), hW, DD, DD);
    gemm(hA, hW, tC, P(sdb), nullptr, nullptr, 1.0f, 0, NN, DD, DD);    // sd
    packA(tB, hA, NN, DD); packB(P(g1w), hW, DD, DD);
    gemm(hA, hW, tD, P(g1b), nullptr, nullptr, 1.0f, 0, NN, DD, DD);    // ga
    packA(tC, hA, NN, DD); packB(P(g2w), hW, DD, DD);
    gemm(hA, hW, tE, P(g2b), nullptr, nullptr, 1.0f, 0, NN, DD, DD);    // gb
    gatefuse_k<<<(NN * DD + 255) / 256, 256, 0, stream>>>(ud, tB, tC, tD, tE, NN * DD);
  };
  inter_half(u1, u2, 41, 42, 37, 38, 29, 30, 31, 32);
  inter_half(u2, u1, 43, 44, 39, 40, 33, 34, 35, 36);
  relu_k<<<(NN * DD + 255) / 256, 256, 0, stream>>>(u1, NN * DD);
  relu_k<<<(NN * DD + 255) / 256, 256, 0, stream>>>(u2, NN * DD);

  // ---- stage 4: m4 refine ----
  auto refine = [&](float* u, const float* sf, int wb, int ww) {
    packBT(sf, hB, DD, NLD);   // B = sf^T  (K=64, Nc=4096)
    packA(u, hA, NN, DD);
    gemm(hA, hB, attL, nullptr, nullptr, nullptr, 1.0f, 0, NN, DD, NLD);
    rowscale_abs_k<<<NN, 256, 0, stream>>>(attL, rsc, NLD, 1.0f / NLD);
    packA(sf, hA, NLD, DD); packB(P(ww), hW, DD, DD);
    gemm(hA, hW, tBig, nullptr, nullptr, nullptr, 1.0f, 0, NLD, DD, DD);  // sf@W
    packA(attL, hA, NN, NLD); packB(tBig, hB, NLD, DD);
    gemm(hA, hB, u, P(wb), u, rsc, 1.0f, 1, NN, NLD, DD);  // relu(u + rsc*(A@w)+b)
  };
  refine(u1, s1f, 45, 46);
  refine(u2, s2f, 47, 48);

  // ---- stage 5: predict ----
  build_x_k<<<2048, 2 * DD, 0, stream>>>(u1, u2, emb_item, i_node, negs, fX);
  packA(fX, hA, 2048, 2 * DD);
  packB(P(50), hW, 2 * DD, HH);
  gemm(hA, hW, Hprd, P(49), nullptr, nullptr, 1.0f, 1, 2048, 2 * DD, HH);
  fc2_sig_k<<<2048, HH, 0, stream>>>(Hprd, P(52), P(51), (float*)d_out);
}
